// ProteinFeatures_12335146074491
// MI455X (gfx1250) — matrix operand order repaired and checked by reading, not compiled, not run on hardware
//
#include <hip/hip_runtime.h>
#include <hip/hip_bf16.h>
#include <math.h>

#define TOPK 30
#define NB 8
#define NL 2048
#define CH 128

typedef _Float16 v16h __attribute__((ext_vector_type(16)));
typedef float    v8f  __attribute__((ext_vector_type(8)));
typedef unsigned int u32x4 __attribute__((ext_vector_type(4)));
typedef int      i32x4 __attribute__((ext_vector_type(4)));
typedef int      i32x8 __attribute__((ext_vector_type(8)));

__device__ __forceinline__ float sgnf(float x) {
    return (x > 0.f) ? 1.f : ((x < 0.f) ? -1.f : 0.f);
}
__device__ __forceinline__ void norm3(float* v) {
    float n = sqrtf(v[0]*v[0] + v[1]*v[1] + v[2]*v[2]);
    n = fmaxf(n, 1e-12f);
    float inv = 1.f / n;
    v[0] *= inv; v[1] *= inv; v[2] *= inv;
}
__device__ __forceinline__ void cross3(const float* a, const float* b, float* c) {
    c[0] = a[1]*b[2] - a[2]*b[1];
    c[1] = a[2]*b[0] - a[0]*b[2];
    c[2] = a[0]*b[1] - a[1]*b[0];
}

// ---------------- Kernel 1: weight precast to f16 (K zero-padded) ----------------
__global__ __launch_bounds__(256) void prep_weights(const float* __restrict__ Wn,
                                                    const float* __restrict__ We,
                                                    _Float16* __restrict__ Wn16,
                                                    _Float16* __restrict__ We16) {
    int idx = blockIdx.x * 256 + threadIdx.x;
    if (idx < 32 * CH) {
        int k = idx >> 7, n = idx & 127;
        Wn16[idx] = (_Float16)((k < 6) ? Wn[k * CH + n] : 0.f);
    }
    if (idx < 64 * CH) {
        int k = idx >> 7, n = idx & 127;
        We16[idx] = (_Float16)((k < 39) ? We[k * CH + n] : 0.f);
    }
}

// ---------------- Kernel 2: per-residue dihedrals + orientation frame ----------------
__global__ __launch_bounds__(128) void node_feat_kernel(const float* __restrict__ X,
                                                        float* __restrict__ Vfeat,
                                                        float* __restrict__ Omat) {
    int idx = blockIdx.x * 128 + threadIdx.x;
    if (idx >= NB * NL) return;
    int b = idx / NL, l = idx % NL;
    const float* Xb = X + (size_t)b * NL * 12;   // [L][4][3]

    float vf[6];
#pragma unroll
    for (int c = 0; c < 3; ++c) {
        float ang = 0.f;
        int q = 3 * l + c - 1;
        if (q >= 0 && q <= 3 * NL - 4) {
            float P[4][3];
#pragma unroll
            for (int s = 0; s < 4; ++s) {
                int t = q + s, r = t / 3, a = t - 3 * r;      // a: 0=N,1=CA,2=C
                const float* p = Xb + (size_t)r * 12 + a * 3;
                P[s][0] = p[0]; P[s][1] = p[1]; P[s][2] = p[2];
            }
            float u2[3], u1[3], u0[3];
#pragma unroll
            for (int d = 0; d < 3; ++d) {
                u2[d] = P[1][d] - P[0][d];
                u1[d] = P[2][d] - P[1][d];
                u0[d] = P[3][d] - P[2][d];
            }
            norm3(u2); norm3(u1); norm3(u0);
            float n2[3], n1[3];
            cross3(u2, u1, n2); cross3(u1, u0, n1);
            norm3(n2); norm3(n1);
            float cosD = n2[0]*n1[0] + n2[1]*n1[1] + n2[2]*n1[2];
            cosD = fminf(fmaxf(cosD, -1.f + 1e-7f), 1.f - 1e-7f);
            float s = sgnf(u2[0]*n1[0] + u2[1]*n1[1] + u2[2]*n1[2]);
            ang = s * acosf(cosD);
        }
        vf[c]     = cosf(ang);
        vf[3 + c] = sinf(ang);
    }
#pragma unroll
    for (int k = 0; k < 6; ++k) Vfeat[(size_t)idx * 6 + k] = vf[k];

    // orientation frame rows [b; n1; x]
    const float* pr = Xb + (size_t)l * 12;
    float nca[3], cac[3];
#pragma unroll
    for (int d = 0; d < 3; ++d) {
        nca[d] = pr[3 + d] - pr[d];       // CA - N
        cac[d] = pr[6 + d] - pr[3 + d];   // C - CA
    }
    norm3(nca); norm3(cac);
    float n1v[3]; cross3(nca, cac, n1v); norm3(n1v);
    float bb[3];
#pragma unroll
    for (int d = 0; d < 3; ++d) bb[d] = cac[d] - nca[d];
    norm3(bb);
    float xv[3]; cross3(bb, n1v, xv); norm3(xv);
    float* O = Omat + (size_t)idx * 9;
    O[0]=bb[0];  O[1]=bb[1];  O[2]=bb[2];
    O[3]=n1v[0]; O[4]=n1v[1]; O[5]=n1v[2];
    O[6]=xv[0];  O[7]=xv[1];  O[8]=xv[2];
}

// ---------------- Kernel 3: KNN (top-30) with LDS candidate tiles ----------------
__global__ __launch_bounds__(128) void knn_kernel(const float* __restrict__ X,
                                                  const float* __restrict__ mask,
                                                  float* __restrict__ Dnb,
                                                  int* __restrict__ Eidx) {
    __shared__ float sx[128], sy[128], sz[128], sm[128];
    int q = blockIdx.x * 128 + threadIdx.x;        // flat query, same batch per block
    int b = q / NL;
    const float* pi = X + ((size_t)q * 4 + 1) * 3; // CA of query
    float xi = pi[0], yi = pi[1], zi = pi[2];
    float mi = mask[q];

    float best[TOPK]; int bidx[TOPK];
#pragma unroll
    for (int s = 0; s < TOPK; ++s) { best[s] = 3e38f; bidx[s] = 0; }
    float wmax = 3e38f; int wslot = 0;

    for (int jt = 0; jt < NL; jt += 128) {
        int j = jt + threadIdx.x;
        const float* pj = X + (((size_t)b * NL + j) * 4 + 1) * 3;
        sx[threadIdx.x] = pj[0]; sy[threadIdx.x] = pj[1]; sz[threadIdx.x] = pj[2];
        sm[threadIdx.x] = mask[(size_t)b * NL + j];
        __syncthreads();
        for (int jj = 0; jj < 128; ++jj) {
            float dx = sx[jj] - xi, dy = sy[jj] - yi, dz = sz[jj] - zi;
            float d = sqrtf(dx*dx + dy*dy + dz*dz + 1e-6f);
            if (mi * sm[jj] == 0.f) d = 1e30f;   // masked pairs pushed to the end
            if (d < wmax) {
                best[wslot] = d; bidx[wslot] = jt + jj;
                wmax = -1.f;
#pragma unroll
                for (int s = 0; s < TOPK; ++s)
                    if (best[s] > wmax) { wmax = best[s]; wslot = s; }
            }
        }
        __syncthreads();
    }
    // ascending selection sort
    for (int a = 0; a < TOPK; ++a) {
        int m2 = a;
        for (int c = a + 1; c < TOPK; ++c) if (best[c] < best[m2]) m2 = c;
        float tv = best[a]; best[a] = best[m2]; best[m2] = tv;
        int   ti = bidx[a]; bidx[a] = bidx[m2]; bidx[m2] = ti;
    }
    for (int a = 0; a < TOPK; ++a) {
        Dnb[(size_t)q * TOPK + a]  = best[a];
        Eidx[(size_t)q * TOPK + a] = bidx[a];
    }
}

// parallel LayerNorm over a 16x128 tile in LDS; 128 threads = 16 rows x 8 chunks
__device__ __forceinline__ void tile_layernorm_store(float (*Ct)[CH], float (*ps)[8],
                                                     int t, const float* g,
                                                     const float* beta,
                                                     float* out_base) {
    int row = t >> 3, chunk = t & 7;
    int c0 = chunk * 16;
    float s = 0.f;
#pragma unroll
    for (int c = 0; c < 16; ++c) s += Ct[row][c0 + c];
    ps[row][chunk] = s;
    __syncthreads();
    float mu = 0.f;
#pragma unroll
    for (int c = 0; c < 8; ++c) mu += ps[row][c];
    mu *= (1.f / CH);
    float v = 0.f;
#pragma unroll
    for (int c = 0; c < 16; ++c) { float d = Ct[row][c0 + c] - mu; v += d * d; }
    __syncthreads();
    ps[row][chunk] = v;
    __syncthreads();
    float var = 0.f;
#pragma unroll
    for (int c = 0; c < 8; ++c) var += ps[row][c];
    var *= (1.f / CH);
    float inv = 1.f / sqrtf(var + 1e-5f);
    float* out = out_base + (size_t)row * CH + c0;
#pragma unroll
    for (int c = 0; c < 16; ++c)
        out[c] = (Ct[row][c0 + c] - mu) * inv * g[c0 + c] + beta[c0 + c];
}

// ---------------- Kernel 4: node GEMM (6->128 via f16 WMMA) + LayerNorm ----------------
__global__ __launch_bounds__(128) void node_gemm_ln(const _Float16* __restrict__ Wn16,
                                                    const float* __restrict__ bn,
                                                    const float* __restrict__ gn,
                                                    const float* __restrict__ betan,
                                                    const float* __restrict__ Vfeat,
                                                    float* __restrict__ Vout) {
    __shared__ _Float16 Ah[16][32];
    __shared__ float Ct[16][CH];
    __shared__ float ps[16][8];
    const int t = threadIdx.x;
    const int lane = t & 31, w = t >> 5;
    const int m = lane & 15, hi = lane >> 4;

    __builtin_prefetch(Wn16 + (size_t)lane * CH, 0, 1);

    for (int idx = t; idx < 16 * 32; idx += 128) {
        int r = idx >> 5, k = idx & 31;
        size_t row = (size_t)blockIdx.x * 16 + r;
        Ah[r][k] = (_Float16)((k < 6) ? Vfeat[row * 6 + k] : 0.f);
    }
    __syncthreads();

    // A fragment: 16-bit A 16x32 layout (lane m, halves -> K = e + 8*hi (+8 for e>=8))
    v16h a;
#pragma unroll
    for (int v = 0; v < 8; ++v) {
        int e = 2 * v;
        int kb = e + hi * 8 + (e >= 8 ? 8 : 0);
        a[e]     = Ah[m][kb];
        a[e + 1] = Ah[m][kb + 1];
    }
#pragma unroll
    for (int nn = 0; nn < 2; ++nn) {
        int n0 = (w * 2 + nn) * 16;
        v16h bfrag;
#pragma unroll
        for (int s = 0; s < 16; ++s)
            bfrag[s] = Wn16[(size_t)(s + hi * 8 + (s >= 8 ? 8 : 0)) * CH + n0 + m];
        v8f c = {};
        c = __builtin_amdgcn_wmma_f32_16x16x32_f16(false, a, false, bfrag,
                                                   (short)0, c, false, false);
#pragma unroll
        for (int r = 0; r < 8; ++r)
            Ct[r + hi * 8][n0 + m] = c[r] + bn[n0 + m];
    }
    __syncthreads();

    tile_layernorm_store(Ct, ps, t, gn, betan,
                         Vout + (size_t)blockIdx.x * 16 * CH);
}

// ---------------- Kernel 5: fused edge features + TDM weight staging + GEMM + LN ----------------
__global__ __launch_bounds__(128) void edge_gemm_ln(const _Float16* __restrict__ We16,
                                                    const float* __restrict__ be,
                                                    const float* __restrict__ ge,
                                                    const float* __restrict__ betae,
                                                    const float* __restrict__ X,
                                                    const float* __restrict__ Omat,
                                                    const float* __restrict__ Dnb,
                                                    const int* __restrict__ Eidx,
                                                    float* __restrict__ Eout) {
    __shared__ _Float16 Wlds[64][CH];   // 16 KB staged via Tensor Data Mover
    __shared__ _Float16 Ah[16][64];
    __shared__ float Ct[16][CH];
    __shared__ float ps[16][8];
    const int t = threadIdx.x;
    const int lane = t & 31, w = t >> 5;
    const int m = lane & 15, hi = lane >> 4;

    // ---- stage the padded f16 weight matrix (64x128 = 16 KB) into LDS via TDM ----
    if (w == 0) {
        unsigned long long ga = (unsigned long long)(uintptr_t)We16;
        unsigned int ldsOff = (unsigned int)(uintptr_t)&Wlds[0][0];  // low 32b = LDS byte addr
        // D# group0: count=1, lds_addr, global_addr[56:0], type=2
        u32x4 g0 = { 1u, ldsOff, (unsigned int)ga,
                     (unsigned int)((ga >> 32) & 0x1FFFFFFull) | 0x80000000u };
        // D# group1: data_size=1 (2B), tensor_dim0=8192, tensor_dim1=1,
        //            tile_dim0=8192, tile_dim1=1, dim0_stride=8192
        i32x8 g1 = { 0x00010000, 0x20000000, 0x00010000, 0x20000000,
                     1, 8192, 0, 0 };
        i32x4 z4 = { 0, 0, 0, 0 };
#if defined(__clang_major__) && (__clang_major__ >= 23)
        i32x8 z8 = { 0, 0, 0, 0, 0, 0, 0, 0 };
        __builtin_amdgcn_tensor_load_to_lds(g0, g1, z4, z4, z8, 0);
#else
        __builtin_amdgcn_tensor_load_to_lds(g0, g1, z4, z4, 0);
#endif
        __builtin_amdgcn_s_wait_tensorcnt(0);
    }

    // ---- edge features: 4 threads per edge (64 threads active) ----
    if (t < 64) {
        int et = t >> 2, part = t & 3;
        size_t e = (size_t)blockIdx.x * 16 + et;
        int b   = (int)(e / ((size_t)NL * TOPK));
        int rem = (int)(e % ((size_t)NL * TOPK));
        int i = rem / TOPK;
        int k = rem % TOPK;
        size_t eio = ((size_t)b * NL + i) * TOPK + k;
        int j = Eidx[eio];
        _Float16* row = &Ah[et][0];

        if (part == 0) {
            // positional encodings (16)
            float dd = (float)(j - i);
            const float lf = 0.575646273248511f;   // ln(10000)/16
#pragma unroll
            for (int f = 0; f < 8; ++f) {
                float ang = dd * expf(-(float)(2 * f) * lf);
                row[f]     = (_Float16)cosf(ang);
                row[8 + f] = (_Float16)sinf(ang);
            }
        } else if (part == 1) {
            // RBF (16): mu = linspace(0,20,16), sigma = 1.25
            float D = Dnb[eio];
#pragma unroll
            for (int r = 0; r < 16; ++r) {
                float mu = (20.f / 15.f) * (float)r;
                float z = (D - mu) * (1.f / 1.25f);
                row[16 + r] = (_Float16)expf(-z * z);
            }
        } else if (part == 2) {
            // dU (3) + K-pad zeros
            const float* Oi = Omat + ((size_t)b * NL + i) * 9;
            const float* Pi = X + (((size_t)b * NL + i) * 4 + 1) * 3;
            const float* Pj = X + (((size_t)b * NL + j) * 4 + 1) * 3;
            float dX[3] = {Pj[0] - Pi[0], Pj[1] - Pi[1], Pj[2] - Pi[2]};
            float u[3];
#pragma unroll
            for (int s = 0; s < 3; ++s)
                u[s] = Oi[s*3+0]*dX[0] + Oi[s*3+1]*dX[1] + Oi[s*3+2]*dX[2];
            norm3(u);
            row[32] = (_Float16)u[0]; row[33] = (_Float16)u[1]; row[34] = (_Float16)u[2];
#pragma unroll
            for (int z = 39; z < 64; ++z) row[z] = (_Float16)0.f;
        } else {
            // quaternion (4) of R = Oi^T Oj
            const float* Oi = Omat + ((size_t)b * NL + i) * 9;
            const float* Oj = Omat + ((size_t)b * NL + j) * 9;
            float R[3][3];
#pragma unroll
            for (int a2 = 0; a2 < 3; ++a2)
#pragma unroll
                for (int m2 = 0; m2 < 3; ++m2)
                    R[a2][m2] = Oi[a2]*Oj[m2] + Oi[3+a2]*Oj[3+m2] + Oi[6+a2]*Oj[6+m2];
            float Rxx = R[0][0], Ryy = R[1][1], Rzz = R[2][2];
            float m0  = 0.5f * sqrtf(fabsf(1.f + Rxx - Ryy - Rzz + 1e-10f));
            float m1  = 0.5f * sqrtf(fabsf(1.f - Rxx + Ryy - Rzz + 1e-10f));
            float m2v = 0.5f * sqrtf(fabsf(1.f - Rxx - Ryy + Rzz + 1e-10f));
            float qx = sgnf(R[2][1] - R[1][2]) * m0;
            float qy = sgnf(R[0][2] - R[2][0]) * m1;
            float qz = sgnf(R[1][0] - R[0][1]) * m2v;
            float qw = 0.5f * sqrtf(fmaxf(0.f, 1.f + Rxx + Ryy + Rzz));
            float inv = 1.f / fmaxf(sqrtf(qx*qx + qy*qy + qz*qz + qw*qw), 1e-12f);
            row[35] = (_Float16)(qx * inv); row[36] = (_Float16)(qy * inv);
            row[37] = (_Float16)(qz * inv); row[38] = (_Float16)(qw * inv);
        }
    }
    __syncthreads();

    // ---- WMMA: A fragments for the two K-tiles ----
    v16h a0, a1;
#pragma unroll
    for (int v = 0; v < 8; ++v) {
        int e2 = 2 * v;
        int kb = e2 + hi * 8 + (e2 >= 8 ? 8 : 0);
        a0[e2]     = Ah[m][kb];
        a0[e2 + 1] = Ah[m][kb + 1];
        a1[e2]     = Ah[m][32 + kb];
        a1[e2 + 1] = Ah[m][32 + kb + 1];
    }
#pragma unroll
    for (int nn = 0; nn < 2; ++nn) {
        int n0 = (w * 2 + nn) * 16;
        v16h b0, b1;
#pragma unroll
        for (int s = 0; s < 16; ++s) {
            int kb = s + hi * 8 + (s >= 8 ? 8 : 0);
            b0[s] = Wlds[kb][n0 + m];
            b1[s] = Wlds[32 + kb][n0 + m];
        }
        v8f c = {};
        c = __builtin_amdgcn_wmma_f32_16x16x32_f16(false, a0, false, b0,
                                                   (short)0, c, false, false);
        c = __builtin_amdgcn_wmma_f32_16x16x32_f16(false, a1, false, b1,
                                                   (short)0, c, false, false);
#pragma unroll
        for (int r = 0; r < 8; ++r)
            Ct[r + hi * 8][n0 + m] = c[r] + be[n0 + m];
    }
    __syncthreads();

    tile_layernorm_store(Ct, ps, t, ge, betae,
                         Eout + (size_t)blockIdx.x * 16 * CH);
}

extern "C" void kernel_launch(void* const* d_in, const int* in_sizes, int n_in,
                              void* d_out, int out_size, void* d_ws, size_t ws_size,
                              hipStream_t stream) {
    (void)in_sizes; (void)n_in; (void)out_size; (void)ws_size;
    const float* X     = (const float*)d_in[0];
    const float* mask  = (const float*)d_in[1];
    const float* Wn    = (const float*)d_in[2];
    const float* bn    = (const float*)d_in[3];
    const float* gn    = (const float*)d_in[4];
    const float* betan = (const float*)d_in[5];
    const float* We    = (const float*)d_in[6];
    const float* be    = (const float*)d_in[7];
    const float* ge    = (const float*)d_in[8];
    const float* betae = (const float*)d_in[9];

    // workspace layout (bytes)
    char* ws = (char*)d_ws;
    _Float16* Wn16 = (_Float16*)(ws + 0);                       //  8 KB
    _Float16* We16 = (_Float16*)(ws + 8192);                    // 16 KB
    float* Vfeat   = (float*)(ws + 24576);                      // B*L*6  = 384 KB
    float* Omat    = (float*)(ws + 24576 + 393216);             // B*L*9  = 576 KB
    float* Dnb     = (float*)(ws + 24576 + 393216 + 589824);    // B*L*30 = 1.9 MB

    float* Vout = (float*)d_out;
    float* Eout = Vout + (size_t)NB * NL * CH;
    int*   Eidx = (int*)(Eout + (size_t)NB * NL * TOPK * CH);

    prep_weights<<<32, 256, 0, stream>>>(Wn, We, Wn16, We16);
    node_feat_kernel<<<(NB * NL) / 128, 128, 0, stream>>>(X, Vfeat, Omat);
    knn_kernel<<<(NB * NL) / 128, 128, 0, stream>>>(X, mask, Dnb, Eidx);
    node_gemm_ln<<<(NB * NL) / 16, 128, 0, stream>>>(Wn16, bn, gn, betan, Vfeat, Vout);
    edge_gemm_ln<<<(NB * NL * TOPK) / 16, 128, 0, stream>>>(We16, be, ge, betae,
                                                            X, Omat, Dnb, Eidx, Eout);
}
